// Block_41781441856223
// MI455X (gfx1250) — matrix operand order repaired and checked by reading, not compiled, not run on hardware
//
#include <hip/hip_runtime.h>
#include <hip/hip_bf16.h>
#include <cstdint>

// ---------------- problem dims ----------------
#define B_  2
#define T_  2048
#define D_  2048
#define N_  16
#define K_  8
#define H_  128
#define F_  8192
#define G_  (N_ / K_)          // 2 query heads per kv head
#define NH_ (N_ * H_)          // 2048
#define KH_ (K_ * H_)          // 1024
#define BT_ ((size_t)B_ * T_)  // 4096

typedef __bf16 bf16;
typedef __attribute__((ext_vector_type(16))) __bf16 bf16x16;
typedef __attribute__((ext_vector_type(8)))  __bf16 bf16x8;
typedef __attribute__((ext_vector_type(8)))  float  f32x8;

union FragBF { bf16x16 f; bf16x8 h[2]; };

__device__ __forceinline__ f32x8 wmma_bf16(bf16x16 a, bf16x16 b, f32x8 c) {
  return __builtin_amdgcn_wmma_f32_16x16x32_bf16(false, a, false, b,
                                                 (short)0, c, false, false);
}

// CDNA5 async global->LDS copy (per-lane, ASYNCcnt-tracked).
// LDS byte offset = low 32 bits of the generic shared-memory address.
__device__ __forceinline__ void async_load_b128(const bf16* sdst, const bf16* gsrc) {
  unsigned lds_off = (unsigned)(size_t)sdst;
  unsigned long long ga = (unsigned long long)gsrc;
  asm volatile("global_load_async_to_lds_b128 %0, %1, off"
               :: "v"(lds_off), "v"(ga) : "memory");
}
__device__ __forceinline__ void wait_asynccnt0() {
  asm volatile("s_wait_asynccnt 0x0" ::: "memory");
}

// ---------------- elementwise kernels ----------------
__global__ void cast_bf16_kernel(const float* __restrict__ in, bf16* __restrict__ out, size_t n) {
  size_t i = (size_t)blockIdx.x * blockDim.x + threadIdx.x;
  if (i < n) out[i] = (bf16)in[i];
}

// generic: in [R][C] f32 row-major -> out [C][R] bf16 (i.e. B^T for GEMM)
__global__ void transpose_cast_kernel(const float* __restrict__ in, bf16* __restrict__ out,
                                      int R, int C) {
  size_t o = (size_t)blockIdx.x * blockDim.x + threadIdx.x;
  if (o >= (size_t)R * C) return;
  int c = (int)(o / R);
  int r = (int)(o % R);
  out[o] = (bf16)in[(size_t)r * C + c];
}

// w_q: (N,D,H) f32 -> Bt layout [NH][D] bf16 : out[(n*H+h)*D + d]
__global__ void permute_wq_kernel(const float* __restrict__ in, bf16* __restrict__ out) {
  size_t o = (size_t)blockIdx.x * blockDim.x + threadIdx.x;
  if (o >= (size_t)NH_ * D_) return;
  int nh = (int)(o / D_);
  int d  = (int)(o % D_);
  int n  = nh / H_;
  int h  = nh % H_;
  out[o] = (bf16)in[((size_t)n * D_ + d) * H_ + h];
}

// w_kv[e]: (K,D,H) f32 -> Bt layout [KH][D] bf16
__global__ void permute_wkv_kernel(const float* __restrict__ in, bf16* __restrict__ out) {
  size_t o = (size_t)blockIdx.x * blockDim.x + threadIdx.x;
  if (o >= (size_t)KH_ * D_) return;
  int kh = (int)(o / D_);
  int d  = (int)(o % D_);
  int k  = kh / H_;
  int h  = kh % H_;
  out[o] = (bf16)in[((size_t)k * D_ + d) * H_ + h];
}

// one row of D per block, 256 threads
__global__ __launch_bounds__(256) void rmsnorm_kernel(const float* __restrict__ x,
                                                      const float* __restrict__ scale,
                                                      bf16* __restrict__ out) {
  const int row = blockIdx.x;
  const float* xr = x + (size_t)row * D_;
  float ss = 0.0f;
  for (int i = threadIdx.x; i < D_; i += 256) { float v = xr[i]; ss += v * v; }
  #pragma unroll
  for (int m = 1; m < 32; m <<= 1) ss += __shfl_xor(ss, m, 32);
  __shared__ float red[8];
  if ((threadIdx.x & 31) == 0) red[threadIdx.x >> 5] = ss;
  __syncthreads();
  float tot = 0.0f;
  #pragma unroll
  for (int i = 0; i < 8; i++) tot += red[i];
  float inv = rsqrtf(tot / (float)D_ + 1e-6f);
  for (int i = threadIdx.x; i < D_; i += 256)
    out[(size_t)row * D_ + i] = (bf16)(xr[i] * inv * (1.0f + scale[i]));
}

// in-place RoPE on bf16 [BT, heads*H]; also applies outscale (H^-0.5 for q)
__global__ void rope_kernel(bf16* __restrict__ q, int heads, float outscale) {
  size_t idx = (size_t)blockIdx.x * blockDim.x + threadIdx.x;
  size_t total = BT_ * (size_t)heads * (H_ / 2);
  if (idx >= total) return;
  int i = (int)(idx & (H_ / 2 - 1));     // 0..63
  size_t rem = idx >> 6;
  int hd = (int)(rem % heads);
  size_t bt = rem / heads;
  int t = (int)(bt % T_);
  float fe = (2.0f / (float)H_) * (float)i;
  float inv_ts = __expf(-fe * __logf(10000.0f));
  float rad = (float)t * inv_ts;
  float s = __sinf(rad), c = __cosf(rad);
  bf16* p = q + bt * (size_t)(heads * H_) + (size_t)hd * H_;
  float x1 = (float)p[i];
  float x2 = (float)p[i + H_ / 2];
  p[i]          = (bf16)((x1 * c - x2 * s) * outscale);
  p[i + H_ / 2] = (bf16)((x2 * c + x1 * s) * outscale);
}

// gate = gelu_tanh(gate) * up   (in place on gate)
__global__ void geglu_kernel(bf16* __restrict__ gate, const bf16* __restrict__ up, size_t n) {
  size_t i = (size_t)blockIdx.x * blockDim.x + threadIdx.x;
  if (i >= n) return;
  float g = (float)gate[i];
  float u = (float)up[i];
  float t = tanhf(0.7978845608028654f * (g + 0.044715f * g * g * g));
  gate[i] = (bf16)(0.5f * g * (1.0f + t) * u);
}

__global__ void add_inplace_kernel(float* __restrict__ a, const float* __restrict__ b, size_t n) {
  size_t i = (size_t)blockIdx.x * blockDim.x + threadIdx.x;
  if (i < n) a[i] += b[i];
}

// ---------------- WMMA GEMM:  C[M,N] = A[M,K] * B[K,N],  B given transposed [N][K] ----------
// A staged to LDS via global_load_async_to_lds_b128; B fragments read straight from global
// (weights are L2-resident: 192 MB L2 vs ~126 MB of bf16 weights).
#define BM 128
#define BN 256
#define BK 32
#define LDA 40

template <bool OUT_BF>
__global__ __launch_bounds__(256) void gemm_bf16_kernel(const bf16* __restrict__ A,
                                                        const bf16* __restrict__ Bt, // [N][Kd]
                                                        void* __restrict__ Cout,
                                                        int M, int N, int Kd) {
  __shared__ __align__(16) bf16 sA[BM * LDA];
  const int tid  = threadIdx.x;
  const int lane = tid & 31;
  const int wave = tid >> 5;
  const int wRow = wave & 1;   // 2 row groups of 64
  const int wCol = wave >> 1;  // 4 col groups of 64
  const int half = lane >> 4;
  const int l16  = lane & 15;
  const int bm = blockIdx.y * BM;
  const int bn = blockIdx.x * BN;

  f32x8 acc[4][4];
  #pragma unroll
  for (int i = 0; i < 4; i++)
    #pragma unroll
    for (int j = 0; j < 4; j++)
      #pragma unroll
      for (int e = 0; e < 8; e++) acc[i][j][e] = 0.0f;

  for (int k0 = 0; k0 < Kd; k0 += BK) {
    // async-stage A tile (BM x BK): 512 x 16B chunks, 2 per thread
    #pragma unroll
    for (int i = 0; i < 2; i++) {
      int c   = tid * 2 + i;
      int row = c >> 2;
      int cc  = c & 3;
      async_load_b128(&sA[row * LDA + cc * 8],
                      A + (size_t)(bm + row) * Kd + k0 + cc * 8);
    }
    if (k0 + BK < Kd)  // global_prefetch_b8 of the next B stripe
      __builtin_prefetch(Bt + (size_t)(bn + (tid & 255)) * Kd + k0 + BK, 0, 0);
    wait_asynccnt0();
    __syncthreads();

    FragBF af[4];
    #pragma unroll
    for (int tr = 0; tr < 4; tr++) {
      const bf16* p = &sA[(wRow * 64 + tr * 16 + l16) * LDA];
      af[tr].h[0] = *(const bf16x8*)(p + half * 16);       // K {0..7} / {16..23}
      af[tr].h[1] = *(const bf16x8*)(p + half * 16 + 8);   // K {8..15} / {24..31}
    }
    FragBF bfg[4];
    #pragma unroll
    for (int tc = 0; tc < 4; tc++) {
      const bf16* p = Bt + (size_t)(bn + wCol * 64 + tc * 16 + l16) * Kd + k0 + half * 16;
      bfg[tc].h[0] = *(const bf16x8*)(p);                  // K {0..15} / {16..31}
      bfg[tc].h[1] = *(const bf16x8*)(p + 8);
    }
    #pragma unroll
    for (int tr = 0; tr < 4; tr++)
      #pragma unroll
      for (int tc = 0; tc < 4; tc++)
        acc[tr][tc] = wmma_bf16(af[tr].f, bfg[tc].f, acc[tr][tc]);
    __syncthreads();
  }

  #pragma unroll
  for (int tr = 0; tr < 4; tr++)
    #pragma unroll
    for (int tc = 0; tc < 4; tc++) {
      int col  = bn + wCol * 64 + tc * 16 + l16;
      int row0 = bm + wRow * 64 + tr * 16 + half * 8;
      #pragma unroll
      for (int r = 0; r < 8; r++) {
        size_t idx = (size_t)(row0 + r) * N + col;
        float v = acc[tr][tc][r];
        if (OUT_BF) ((bf16*)Cout)[idx] = (bf16)v;
        else        ((float*)Cout)[idx] = v;
      }
    }
}

// ---------------- flash attention (causal, GQA), WMMA for QK^T and PV ----------------
#define TQ 64  // query rows per block (4 waves x 16)
#define SC 32  // key chunk

__global__ __launch_bounds__(128) void attn_kernel(const bf16* __restrict__ qb,
                                                   const bf16* __restrict__ kb,
                                                   const bf16* __restrict__ vb,
                                                   bf16* __restrict__ enc) {
  const int b   = blockIdx.z;
  const int n   = blockIdx.y;
  const int q0  = blockIdx.x * TQ;
  const int kh  = n / G_;
  const int tid = threadIdx.x, lane = tid & 31, wave = tid >> 5;
  const int half = lane >> 4, l16 = lane & 15;

  __shared__ __align__(16) bf16 Vt[H_ * 40];       // [h][key], stride 40
  __shared__ __align__(16) bf16 Pw[4][16 * 40];    // per-wave P [row][key]

  // load q fragments once: 16 rows per wave, H=128 -> 4 K-slices of 32
  FragBF aq[4];
  {
    int row = q0 + wave * 16 + l16;
    const bf16* qp = qb + ((size_t)(b * T_ + row)) * NH_ + (size_t)n * H_;
    #pragma unroll
    for (int t = 0; t < 4; t++) {
      aq[t].h[0] = *(const bf16x8*)(qp + t * 32 + half * 16);
      aq[t].h[1] = *(const bf16x8*)(qp + t * 32 + half * 16 + 8);
    }
  }

  f32x8 acc[8];
  #pragma unroll
  for (int t = 0; t < 8; t++)
    #pragma unroll
    for (int e = 0; e < 8; e++) acc[t][e] = 0.0f;
  float mrun[8], lrun[8];
  #pragma unroll
  for (int r = 0; r < 8; r++) { mrun[r] = -3.0e38f; lrun[r] = 0.0f; }

  const int qrow_base = q0 + wave * 16 + half * 8;  // acc row r -> query qrow_base + r
  const float BIG_NEG = -2.3819763e38f;
  const int smax = q0 + TQ;

  for (int s0 = 0; s0 < smax; s0 += SC) {
    // cooperative V chunk load -> transposed LDS [h][key]
    #pragma unroll
    for (int i = 0; i < 4; i++) {
      int c   = tid * 4 + i;     // 0..511
      int key = c >> 4;          // 0..31
      int hc  = c & 15;          // h = hc*8
      bf16x8 v = *(const bf16x8*)(vb + ((size_t)(b * T_ + s0 + key)) * KH_ +
                                  (size_t)kh * H_ + hc * 8);
      #pragma unroll
      for (int j = 0; j < 8; j++) Vt[(hc * 8 + j) * 40 + key] = v[j];
    }

    // logits: q (16xH) x K^T (Hx32) -> two 16x16 tiles
    f32x8 e0, e1;
    #pragma unroll
    for (int i = 0; i < 8; i++) { e0[i] = 0.0f; e1[i] = 0.0f; }
    const bf16* kbase = kb + ((size_t)(b * T_ + s0)) * KH_ + (size_t)kh * H_;
    #pragma unroll
    for (int t = 0; t < 4; t++) {
      FragBF bk0, bk1;
      const bf16* kp0 = kbase + (size_t)l16 * KH_;
      const bf16* kp1 = kbase + (size_t)(16 + l16) * KH_;
      bk0.h[0] = *(const bf16x8*)(kp0 + t * 32 + half * 16);
      bk0.h[1] = *(const bf16x8*)(kp0 + t * 32 + half * 16 + 8);
      bk1.h[0] = *(const bf16x8*)(kp1 + t * 32 + half * 16);
      bk1.h[1] = *(const bf16x8*)(kp1 + t * 32 + half * 16 + 8);
      e0 = wmma_bf16(aq[t].f, bk0.f, e0);
      e1 = wmma_bf16(aq[t].f, bk1.f, e1);
    }

    // causal mask + online softmax (row stats across 16-lane halves)
    int kg0 = s0 + l16, kg1 = kg0 + 16;
    #pragma unroll
    for (int r = 0; r < 8; r++) {
      int qg = qrow_base + r;
      float v0 = (kg0 <= qg) ? e0[r] : BIG_NEG;
      float v1 = (kg1 <= qg) ? e1[r] : BIG_NEG;
      float mx = fmaxf(v0, v1);
      #pragma unroll
      for (int m = 1; m < 16; m <<= 1) mx = fmaxf(mx, __shfl_xor(mx, m, 32));
      float mnew = fmaxf(mrun[r], mx);
      float corr = __expf(mrun[r] - mnew);
      float p0 = __expf(v0 - mnew);
      float p1 = __expf(v1 - mnew);
      float rs = p0 + p1;
      #pragma unroll
      for (int m = 1; m < 16; m <<= 1) rs += __shfl_xor(rs, m, 32);
      lrun[r] = lrun[r] * corr + rs;
      mrun[r] = mnew;
      #pragma unroll
      for (int t = 0; t < 8; t++) acc[t][r] *= corr;
      int prow = half * 8 + r;
      Pw[wave][prow * 40 + l16]      = (bf16)p0;
      Pw[wave][prow * 40 + 16 + l16] = (bf16)p1;
    }
    __syncthreads();

    // O += P (16x32) x V (32xH)
    FragBF ap;
    ap.h[0] = *(const bf16x8*)(&Pw[wave][l16 * 40 + half * 16]);
    ap.h[1] = *(const bf16x8*)(&Pw[wave][l16 * 40 + half * 16 + 8]);
    #pragma unroll
    for (int t = 0; t < 8; t++) {
      FragBF bv;
      const bf16* vp = &Vt[(t * 16 + l16) * 40 + half * 16];
      bv.h[0] = *(const bf16x8*)(vp);
      bv.h[1] = *(const bf16x8*)(vp + 8);
      acc[t] = wmma_bf16(ap.f, bv.f, acc[t]);
    }
    __syncthreads();
  }

  // normalize and write enc [BT, N*H] bf16
  #pragma unroll
  for (int r = 0; r < 8; r++) {
    float inv = 1.0f / lrun[r];
    int qg = qrow_base + r;
    bf16* op = enc + ((size_t)(b * T_ + qg)) * NH_ + (size_t)n * H_;
    #pragma unroll
    for (int t = 0; t < 8; t++) op[t * 16 + l16] = (bf16)(acc[t][r] * inv);
  }
}

// ---------------- host launcher ----------------
extern "C" void kernel_launch(void* const* d_in, const int* in_sizes, int n_in,
                              void* d_out, int out_size, void* d_ws, size_t ws_size,
                              hipStream_t stream) {
  (void)in_sizes; (void)n_in; (void)out_size; (void)ws_size;
  const float* x      = (const float*)d_in[0];
  // d_in[1] = positions (arange, computed analytically), d_in[2] = causal mask (implicit)
  const float* w_q    = (const float*)d_in[3];
  const float* w_kv   = (const float*)d_in[4];
  const float* w_av   = (const float*)d_in[5];
  const float* s_attn = (const float*)d_in[6];
  const float* s_ffw  = (const float*)d_in[7];
  const float* w_gate = (const float*)d_in[8];
  const float* w_lin  = (const float*)d_in[9];

  char* base = (char*)d_ws;
  size_t off = 0;
  auto carve = [&](size_t bytes) -> char* {
    char* p = base + off;
    off = (off + bytes + 255) & ~(size_t)255;
    return p;
  };
  bf16*  xn    = (bf16*)carve(BT_ * D_ * 2);
  bf16*  wqb   = (bf16*)carve((size_t)NH_ * D_ * 2);   // [NH][D]
  bf16*  wkb   = (bf16*)carve((size_t)KH_ * D_ * 2);   // [KH][D]
  bf16*  wvb   = (bf16*)carve((size_t)KH_ * D_ * 2);   // [KH][D]
  bf16*  wavb  = (bf16*)carve((size_t)D_ * NH_ * 2);   // [D][NH]
  bf16*  wg0b  = (bf16*)carve((size_t)F_ * D_ * 2);    // [F][D]
  bf16*  wg1b  = (bf16*)carve((size_t)F_ * D_ * 2);    // [F][D]
  bf16*  wlb   = (bf16*)carve((size_t)D_ * F_ * 2);    // [D][F]
  bf16*  qbuf  = (bf16*)carve(BT_ * NH_ * 2);
  bf16*  kbuf  = (bf16*)carve(BT_ * KH_ * 2);
  bf16*  vbuf  = (bf16*)carve(BT_ * KH_ * 2);
  bf16*  encb  = (bf16*)carve(BT_ * NH_ * 2);
  float* resid = (float*)carve(BT_ * D_ * 4);
  bf16*  hbuf  = (bf16*)carve(BT_ * D_ * 2);
  bf16*  gate  = (bf16*)carve(BT_ * F_ * 2);
  bf16*  upb   = (bf16*)carve(BT_ * F_ * 2);

  auto nblk = [](size_t n, int t) { return (unsigned)((n + (size_t)t - 1) / (size_t)t); };

  // --- weight conversion to transposed bf16 layouts [cols][K] ---
  transpose_cast_kernel<<<nblk((size_t)NH_ * D_, 256), 256, 0, stream>>>(w_av, wavb, NH_, D_);
  transpose_cast_kernel<<<nblk((size_t)D_ * F_, 256), 256, 0, stream>>>(w_gate, wg0b, D_, F_);
  transpose_cast_kernel<<<nblk((size_t)D_ * F_, 256), 256, 0, stream>>>(w_gate + (size_t)D_ * F_, wg1b, D_, F_);
  transpose_cast_kernel<<<nblk((size_t)F_ * D_, 256), 256, 0, stream>>>(w_lin, wlb, F_, D_);
  permute_wq_kernel<<<nblk((size_t)NH_ * D_, 256), 256, 0, stream>>>(w_q, wqb);
  permute_wkv_kernel<<<nblk((size_t)KH_ * D_, 256), 256, 0, stream>>>(w_kv, wkb);
  permute_wkv_kernel<<<nblk((size_t)KH_ * D_, 256), 256, 0, stream>>>(w_kv + (size_t)K_ * D_ * H_, wvb);

  // --- pre-attn RMSNorm ---
  rmsnorm_kernel<<<(unsigned)BT_, 256, 0, stream>>>(x, s_attn, xn);

  // --- QKV projections (WMMA GEMMs, bf16 out) ---
  gemm_bf16_kernel<true><<<dim3(NH_ / BN, (unsigned)(BT_ / BM)), 256, 0, stream>>>(xn, wqb, qbuf, (int)BT_, NH_, D_);
  gemm_bf16_kernel<true><<<dim3(KH_ / BN, (unsigned)(BT_ / BM)), 256, 0, stream>>>(xn, wkb, kbuf, (int)BT_, KH_, D_);
  gemm_bf16_kernel<true><<<dim3(KH_ / BN, (unsigned)(BT_ / BM)), 256, 0, stream>>>(xn, wvb, vbuf, (int)BT_, KH_, D_);

  // --- RoPE (q also scaled by H^-1/2) ---
  rope_kernel<<<nblk(BT_ * N_ * (H_ / 2), 256), 256, 0, stream>>>(qbuf, N_, 0.08838834764831845f);
  rope_kernel<<<nblk(BT_ * K_ * (H_ / 2), 256), 256, 0, stream>>>(kbuf, K_, 1.0f);

  // --- causal GQA flash attention ---
  attn_kernel<<<dim3(T_ / TQ, N_, B_), 128, 0, stream>>>(qbuf, kbuf, vbuf, encb);

  // --- attention output projection + residual ---
  gemm_bf16_kernel<false><<<dim3(D_ / BN, (unsigned)(BT_ / BM)), 256, 0, stream>>>(encb, wavb, resid, (int)BT_, D_, NH_);
  add_inplace_kernel<<<nblk(BT_ * D_, 256), 256, 0, stream>>>(resid, x, BT_ * D_);

  // --- FFN ---
  rmsnorm_kernel<<<(unsigned)BT_, 256, 0, stream>>>(resid, s_ffw, hbuf);
  gemm_bf16_kernel<true><<<dim3(F_ / BN, (unsigned)(BT_ / BM)), 256, 0, stream>>>(hbuf, wg0b, gate, (int)BT_, F_, D_);
  gemm_bf16_kernel<true><<<dim3(F_ / BN, (unsigned)(BT_ / BM)), 256, 0, stream>>>(hbuf, wg1b, upb, (int)BT_, F_, D_);
  geglu_kernel<<<nblk(BT_ * F_, 256), 256, 0, stream>>>(gate, upb, BT_ * F_);
  gemm_bf16_kernel<false><<<dim3(D_ / BN, (unsigned)(BT_ / BM)), 256, 0, stream>>>(gate, wlb, d_out, (int)BT_, D_, F_);
  add_inplace_kernel<<<nblk(BT_ * D_, 256), 256, 0, stream>>>((float*)d_out, resid, BT_ * D_);
}
